// MultiHeadLatentAttention_47871705481353
// MI455X (gfx1250) — matrix operand order repaired and checked by reading, not compiled, not run on hardware
//
#include <hip/hip_runtime.h>
#include <hip/hip_bf16.h>

typedef __attribute__((ext_vector_type(16))) __bf16        v16bf;
typedef __attribute__((ext_vector_type(8)))  float         v8f;
typedef __attribute__((ext_vector_type(4)))  unsigned int  v4u;
typedef __attribute__((ext_vector_type(8)))  int           v8i;
typedef __attribute__((ext_vector_type(4)))  int           v4i;

__device__ __forceinline__ v8f wmma_bf16(v16bf a, v16bf b, v8f c) {
    return __builtin_amdgcn_wmma_f32_16x16x32_bf16(
        false, a, false, b, (short)0, c, false, false);
}

template<int MASK>
__device__ __forceinline__ float swz_xor(float x) {
    int r = __builtin_amdgcn_ds_swizzle(__float_as_int(x), (MASK << 10) | 0x1F);
    return __int_as_float(r);
}

// ---------------------------------------------------------------------------
// Tensor Data Mover: 2D bf16 tile copy global -> LDS.
//  tile_d0 elements per row (contiguous), tile_d1 rows, row stride d0_stride
//  elements. Optional LDS padding: pad_en inserts 4 DWORDs (8 bf16) after
//  every 16 DWORDs (32 bf16) stored -> matches [rows][32+8] LDS layouts.
// ---------------------------------------------------------------------------
__device__ __forceinline__ void tdm_load_2d_bf16(
    unsigned int lds_byte_off, const void* gptr,
    unsigned int tile_d0, unsigned int tile_d1,
    unsigned long long d0_stride_elems, int pad_en)
{
    unsigned long long ga = (unsigned long long)gptr;
    v4u g0;
    g0[0] = 1u;                                            // count=1
    g0[1] = lds_byte_off;                                  // lds_addr
    g0[2] = (unsigned int)(ga & 0xffffffffu);              // global_addr lo
    g0[3] = (unsigned int)((ga >> 32) & 0x01ffffffu) | (2u << 30); // hi | type=2

    unsigned int td0 = tile_d0, td1 = tile_d1;             // tensor dims = tile dims
    v8i g1;
    g1[0] = (int)((1u << 16)                               // data_size = 2 bytes
                | ((unsigned)(pad_en ? 1u : 0u) << 20)     // pad_enable
                | ((pad_en ? 3u : 0u) << 22)               // pad_interval: 16 DW
                | ((pad_en ? 3u : 0u) << 25));             // pad_amount: 4 DW
    g1[1] = (int)((td0 & 0xffffu) << 16);                  // tensor_dim0[15:0]
    g1[2] = (int)((td0 >> 16) | ((td1 & 0xffffu) << 16));  // dim0[31:16]|dim1[15:0]
    g1[3] = (int)((td1 >> 16) | (tile_d0 << 16));          // dim1[31:16]|tile_dim0
    g1[4] = (int)(tile_d1 & 0xffffu);                      // tile_dim1 (tile_dim2=0)
    g1[5] = (int)(d0_stride_elems & 0xffffffffu);          // dim0_stride[31:0]
    g1[6] = (int)((d0_stride_elems >> 32) & 0xffffu);      // dim0_stride[47:32]
    g1[7] = 0;

    v4i z4 = {0, 0, 0, 0};
#if __has_include(<hip/amd_detail/amd_gfx1250_TDM.h>)
    v8i z8 = {0, 0, 0, 0, 0, 0, 0, 0};
    __builtin_amdgcn_tensor_load_to_lds(g0, g1, z4, z4, z8, 0);
#else
    __builtin_amdgcn_tensor_load_to_lds(g0, g1, z4, z4, 0);
#endif
}

__device__ __forceinline__ unsigned int lds_off(const void* p) {
    // generic LDS pointer: low 32 bits are the LDS byte address
    return (unsigned int)(unsigned long long)p;
}

// ---------------------------------------------------------------------------
// GEMM: C[M,N] = A[M,K](bf16) @ Wt[N,K](bf16, pre-transposed) + bias[N](f32)
// Output bf16 (store_f32==0) or f32 (store_f32!=0).
// Block tile 256x128x32, 8 waves (4x2), 4x4 WMMA tiles/wave.
// TDM double-buffered staging: waves only do ds_load_b128 + WMMA.
// ---------------------------------------------------------------------------
#define GBM 256
#define GBN 128
#define GBK 32
#define KPAD 8

__global__ __launch_bounds__(256)
void gemm_bias_kernel(const __bf16* __restrict__ A, const __bf16* __restrict__ Wt,
                      const float* __restrict__ bias, void* __restrict__ C,
                      int M, int N, int K, int store_f32)
{
    __shared__ __bf16 As[2][GBM][GBK + KPAD];   // [m][k]
    __shared__ __bf16 Bt[2][GBN][GBK + KPAD];   // [n][k]

    const int tid  = threadIdx.x;
    const int wid  = tid >> 5;
    const int lane = tid & 31;
    const int wm   = wid >> 1;       // 0..3 -> 64 rows
    const int wn   = wid & 1;        // 0..1 -> 64 cols
    const int half = lane >> 4;
    const int l16  = lane & 15;
    const int m0   = blockIdx.y * GBM;
    const int n0   = blockIdx.x * GBN;
    const unsigned int brows = (unsigned int)((N - n0 < GBN) ? (N - n0) : GBN);

    v8f acc[4][4] = {};

    auto issue = [&](int buf, int kt) {
        tdm_load_2d_bf16(lds_off(&As[buf][0][0]), A  + (size_t)m0 * K + kt,
                         GBK, GBM, (unsigned long long)K, 1);
        tdm_load_2d_bf16(lds_off(&Bt[buf][0][0]), Wt + (size_t)n0 * K + kt,
                         GBK, brows, (unsigned long long)K, 1);
    };

    if (tid == 0) issue(0, 0);

    int ib = 0;
    for (int kt = 0; kt < K; kt += GBK, ib ^= 1) {
        const bool has_next = (kt + GBK) < K;
        if (tid == 0) {
            if (has_next) {
                issue(ib ^ 1, kt + GBK);
                __builtin_amdgcn_s_wait_tensorcnt(2);
            } else {
                __builtin_amdgcn_s_wait_tensorcnt(0);
            }
        }
        __syncthreads();   // publish staged tiles

        v16bf afr[4], bfr[4];
        #pragma unroll
        for (int i = 0; i < 4; i++) {
            const __bf16* ar = &As[ib][wm * 64 + i * 16 + l16][half * 8];
            #pragma unroll
            for (int j = 0; j < 8; j++) { afr[i][j] = ar[j]; afr[i][8 + j] = ar[16 + j]; }
        }
        #pragma unroll
        for (int jt = 0; jt < 4; jt++) {
            const __bf16* br = &Bt[ib][wn * 64 + jt * 16 + l16][half * 8];
            #pragma unroll
            for (int j = 0; j < 8; j++) { bfr[jt][j] = br[j]; bfr[jt][8 + j] = br[16 + j]; }
        }
        #pragma unroll
        for (int i = 0; i < 4; i++)
            #pragma unroll
            for (int jt = 0; jt < 4; jt++)
                acc[i][jt] = wmma_bf16(afr[i], bfr[jt], acc[i][jt]);
        __syncthreads();   // all reads done before this buffer is re-filled
    }

    #pragma unroll
    for (int i = 0; i < 4; i++)
        #pragma unroll
        for (int jt = 0; jt < 4; jt++) {
            int col = n0 + wn * 64 + jt * 16 + l16;
            if (col < N) {
                float bv = bias[col];
                #pragma unroll
                for (int r = 0; r < 8; r++) {
                    int row = m0 + wm * 64 + i * 16 + r + 8 * half;
                    float val = acc[i][jt][r] + bv;
                    if (store_f32) ((float*)C)[(size_t)row * N + col] = val;
                    else           ((__bf16*)C)[(size_t)row * N + col] = (__bf16)val;
                }
            }
        }
}

// ---------------------------------------------------------------------------
// Prep: fp32 -> bf16 convert, and weight convert+transpose W[K,N]->Wt[N,K].
// ---------------------------------------------------------------------------
__global__ void cvt_bf16_kernel(const float* __restrict__ src, __bf16* __restrict__ dst, long n)
{
    long i = (long)blockIdx.x * blockDim.x + threadIdx.x;
    if (i < n) dst[i] = (__bf16)src[i];
}

__global__ void transpose_w_kernel(const float* __restrict__ W, __bf16* __restrict__ Wt,
                                   int K, int N)
{
    long idx = (long)blockIdx.x * blockDim.x + threadIdx.x;
    if (idx >= (long)K * N) return;
    int k = (int)(idx % K);
    int n = (int)(idx / K);
    Wt[idx] = (__bf16)W[(size_t)k * N + n];
}

// ---------------------------------------------------------------------------
// RoPE in place (bf16 buffer [B*S, pairs_per_row*2]).
// ---------------------------------------------------------------------------
__global__ void rope_kernel(__bf16* __restrict__ x, long total_pairs, int S, int pairs_per_row)
{
    long idx = (long)blockIdx.x * blockDim.x + threadIdx.x;
    if (idx >= total_pairs) return;
    long row = idx / pairs_per_row;
    int  p   = (int)(idx % pairs_per_row);
    int  i   = p & 31;
    int  s   = (int)(row % S);
    float inv = __powf(10000.0f, -(float)(2 * i) / 64.0f);
    float sn, cs;
    __sincosf((float)s * inv, &sn, &cs);
    __bf16* xr = x + row * (long)(pairs_per_row * 2) + p * 2;
    float x0 = (float)xr[0], x1 = (float)xr[1];
    xr[0] = (__bf16)(x0 * cs - x1 * sn);
    xr[1] = (__bf16)(x0 * sn + x1 * cs);
}

// ---------------------------------------------------------------------------
// Packing: q_full/k_full [B,H,S,192] bf16; v_t [B,H,128,S] bf16.
// ---------------------------------------------------------------------------
__global__ void pack_qfull_kernel(const __bf16* __restrict__ q, const __bf16* __restrict__ qrot,
                                  __bf16* __restrict__ qf, int S, int H, long total)
{
    long idx = (long)blockIdx.x * blockDim.x + threadIdx.x;
    if (idx >= total) return;
    int d = (int)(idx % 192);
    long t = idx / 192;
    int s = (int)(t % S); t /= S;
    int h = (int)(t % H); t /= H;
    int b = (int)t;
    long bs = (long)b * S + s;
    float val = (d < 128) ? (float)q[bs * 2048 + h * 128 + d]
                          : (float)qrot[bs * (long)(H * 64) + h * 64 + (d - 128)];
    qf[idx] = (__bf16)(val * 0.07216878364870322f);   // 1/sqrt(192)
}

__global__ void pack_kfull_kernel(const __bf16* __restrict__ k, const __bf16* __restrict__ krot,
                                  __bf16* __restrict__ kf, int S, int H, long total)
{
    long idx = (long)blockIdx.x * blockDim.x + threadIdx.x;
    if (idx >= total) return;
    int d = (int)(idx % 192);
    long t = idx / 192;
    int s = (int)(t % S); t /= S;
    int h = (int)(t % H); t /= H;
    int b = (int)t;
    long bs = (long)b * S + s;
    kf[idx] = (d < 128) ? k[bs * 2048 + h * 128 + d]
                        : krot[bs * 64 + (d - 128)];
}

__global__ void pack_vt_kernel(const __bf16* __restrict__ v, __bf16* __restrict__ vt,
                               int S, int H, long total)
{
    long idx = (long)blockIdx.x * blockDim.x + threadIdx.x;
    if (idx >= total) return;
    int s = (int)(idx % S);
    long t = idx / S;
    int dv = (int)(t % 128); t /= 128;
    int h  = (int)(t % H);   t /= H;
    int b  = (int)t;
    vt[idx] = v[((long)b * S + s) * 2048 + h * 128 + dv];
}

// ---------------------------------------------------------------------------
// Flash attention: qf/kf [B,H,S,192] bf16, vt [B,H,128,S] bf16, o bf16.
// 8 waves/block, 16 query rows/wave, 32-key blocks staged by TDM.
// ---------------------------------------------------------------------------
__global__ __launch_bounds__(256)
void mla_flash_attn(const __bf16* __restrict__ qf, const __bf16* __restrict__ kf,
                    const __bf16* __restrict__ vt, __bf16* __restrict__ o,
                    int S, int H)
{
    const int b = blockIdx.z, h = blockIdx.y;
    const int tid  = threadIdx.x;
    const int wid  = tid >> 5;
    const int lane = tid & 31;
    const int half = lane >> 4;
    const int l16  = lane & 15;
    const int qbase = blockIdx.x * 128 + wid * 16;

    const __bf16* Q  = qf + ((size_t)(b * H + h) * S + qbase) * 192;
    const __bf16* K  = kf + (size_t)(b * H + h) * S * 192;
    const __bf16* Vt = vt + (size_t)(b * H + h) * 128 * S;

    __shared__ __bf16 Ks[32][192];           // [key][d] (no pad)
    __shared__ __bf16 Vs[128][GBK + KPAD];   // [dv][key]
    __shared__ __bf16 plds[8][16 * 32];
    __bf16* pbuf = &plds[wid][0];

    // preload Q A-fragments (d = 6 x 32)
    v16bf qfrag[6];
    {
        const __bf16* qr0 = Q + (size_t)l16 * 192 + half * 8;
        #pragma unroll
        for (int kk = 0; kk < 6; kk++) {
            const __bf16* qr = qr0 + kk * 32;
            #pragma unroll
            for (int j = 0; j < 8; j++) { qfrag[kk][j] = qr[j]; qfrag[kk][8 + j] = qr[16 + j]; }
        }
    }

    v8f acc[8] = {};
    float m8[8], l8[8];
    #pragma unroll
    for (int r = 0; r < 8; r++) { m8[r] = -1e30f; l8[r] = 0.0f; }

    for (int kb = 0; kb < S; kb += 32) {
        __syncthreads();   // previous block's fragment reads complete
        if (tid == 0) {
            tdm_load_2d_bf16(lds_off(&Ks[0][0]), K + (size_t)kb * 192,
                             192, 32, 192ull, 0);
            tdm_load_2d_bf16(lds_off(&Vs[0][0]), Vt + kb,
                             32, 128, (unsigned long long)S, 1);
            __builtin_amdgcn_s_wait_tensorcnt(0);
        }
        __syncthreads();

        // ---- scores ------------------------------------------------------
        v8f stile[2];
        #pragma unroll
        for (int t = 0; t < 2; t++) {
            v8f sacc = {};
            #pragma unroll
            for (int kk = 0; kk < 6; kk++) {
                v16bf bfr;
                const __bf16* kr = &Ks[t * 16 + l16][kk * 32 + half * 8];
                #pragma unroll
                for (int j = 0; j < 8; j++) { bfr[j] = kr[j]; bfr[8 + j] = kr[16 + j]; }
                sacc = wmma_bf16(qfrag[kk], bfr, sacc);
            }
            stile[t] = sacc;
        }

        // ---- online softmax ---------------------------------------------
        float mnew[8], alpha[8], lsum[8];
        #pragma unroll
        for (int r = 0; r < 8; r++) {
            float mx = fmaxf(stile[0][r], stile[1][r]);
            mx = fmaxf(mx, swz_xor<1>(mx));
            mx = fmaxf(mx, swz_xor<2>(mx));
            mx = fmaxf(mx, swz_xor<4>(mx));
            mx = fmaxf(mx, swz_xor<8>(mx));
            mnew[r]  = fmaxf(m8[r], mx);
            alpha[r] = __expf(m8[r] - mnew[r]);
            m8[r]    = mnew[r];
            lsum[r]  = 0.0f;
        }
        #pragma unroll
        for (int t = 0; t < 2; t++)
            #pragma unroll
            for (int r = 0; r < 8; r++) {
                float p = __expf(stile[t][r] - mnew[r]);
                lsum[r] += p;
                pbuf[(r + 8 * half) * 32 + t * 16 + l16] = (__bf16)p;
            }
        #pragma unroll
        for (int r = 0; r < 8; r++) {
            float sum = lsum[r];
            sum += swz_xor<1>(sum);
            sum += swz_xor<2>(sum);
            sum += swz_xor<4>(sum);
            sum += swz_xor<8>(sum);
            l8[r] = l8[r] * alpha[r] + sum;
        }
        #pragma unroll
        for (int nt = 0; nt < 8; nt++)
            #pragma unroll
            for (int r = 0; r < 8; r++) acc[nt][r] *= alpha[r];

        asm volatile("s_wait_dscnt 0x0" ::: "memory");   // wave-local P visibility

        v16bf pfrag;
        {
            const __bf16* pr = pbuf + l16 * 32 + half * 8;
            #pragma unroll
            for (int j = 0; j < 8; j++) { pfrag[j] = pr[j]; pfrag[8 + j] = pr[16 + j]; }
        }
        #pragma unroll
        for (int nt = 0; nt < 8; nt++) {
            v16bf vfr;
            const __bf16* vr = &Vs[nt * 16 + l16][half * 8];
            #pragma unroll
            for (int j = 0; j < 8; j++) { vfr[j] = vr[j]; vfr[8 + j] = vr[16 + j]; }
            acc[nt] = wmma_bf16(pfrag, vfr, acc[nt]);
        }
    }

    #pragma unroll
    for (int nt = 0; nt < 8; nt++) {
        int col = nt * 16 + l16;
        #pragma unroll
        for (int r = 0; r < 8; r++) {
            int row = qbase + r + 8 * half;
            o[((size_t)b * S + row) * 2048 + h * 128 + col] =
                (__bf16)(acc[nt][r] * __frcp_rn(l8[r]));
        }
    }
}

// ---------------------------------------------------------------------------
extern "C" void kernel_launch(void* const* d_in, const int* in_sizes, int n_in,
                              void* d_out, int out_size, void* d_ws, size_t ws_size,
                              hipStream_t stream)
{
    const float* seq   = (const float*)d_in[0];
    const float* W_dkv = (const float*)d_in[1];
    const float* b_dkv = (const float*)d_in[2];
    const float* W_dq  = (const float*)d_in[3];
    const float* b_dq  = (const float*)d_in[4];
    const float* W_uq  = (const float*)d_in[5];
    const float* b_uq  = (const float*)d_in[6];
    const float* W_uk  = (const float*)d_in[7];
    const float* b_uk  = (const float*)d_in[8];
    const float* W_uv  = (const float*)d_in[9];
    const float* b_uv  = (const float*)d_in[10];
    const float* W_rq  = (const float*)d_in[11];
    const float* b_rq  = (const float*)d_in[12];
    const float* W_rk  = (const float*)d_in[13];
    const float* b_rk  = (const float*)d_in[14];
    const float* W_o   = (const float*)d_in[15];
    const float* b_o   = (const float*)d_in[16];

    const int Bc = 4, Sc = 2048, Dm = 2048, Hc = 16, Dc = 512, Dc1 = 1536, Drot = 64;
    const int M = Bc * Sc;

    __bf16* ws = (__bf16*)d_ws;
    size_t off = 0;
    auto alloc = [&](size_t n) { __bf16* p = ws + off; off += n; return p; };

    __bf16* seqb   = alloc((size_t)M * Dm);
    __bf16* Wt_dq  = alloc((size_t)Dc1 * Dm);
    __bf16* Wt_dkv = alloc((size_t)Dc * Dm);
    __bf16* Wt_rk  = alloc((size_t)Drot * Dm);
    __bf16* Wt_uq  = alloc((size_t)Dm * Dc1);
    __bf16* Wt_rq  = alloc((size_t)(Hc * Drot) * Dc1);
    __bf16* Wt_uk  = alloc((size_t)Dm * Dc);
    __bf16* Wt_uv  = alloc((size_t)Dm * Dc);
    __bf16* Wt_o   = alloc((size_t)Dm * Dm);
    __bf16* c_q    = alloc((size_t)M * Dc1);
    __bf16* c_kv   = alloc((size_t)M * Dc);
    __bf16* q      = alloc((size_t)M * Dm);
    __bf16* q_rot  = alloc((size_t)M * Hc * Drot);
    __bf16* k      = alloc((size_t)M * Dm);
    __bf16* vv     = alloc((size_t)M * Dm);
    __bf16* k_rot  = alloc((size_t)M * Drot);
    __bf16* qfull  = alloc((size_t)Bc * Hc * Sc * 192);
    __bf16* kfull  = alloc((size_t)Bc * Hc * Sc * 192);
    __bf16* vt     = alloc((size_t)Bc * Hc * 128 * Sc);
    __bf16* attn   = q;   // q is dead once qfull is built

    auto cvt = [&](const float* s, __bf16* d, long n) {
        cvt_bf16_kernel<<<(int)((n + 255) / 256), 256, 0, stream>>>(s, d, n);
    };
    auto trw = [&](const float* W, __bf16* Wt, int K, int N) {
        long n = (long)K * N;
        transpose_w_kernel<<<(int)((n + 255) / 256), 256, 0, stream>>>(W, Wt, K, N);
    };
    auto gemm = [&](const __bf16* A, const __bf16* Wt, const float* bias, void* C,
                    int m, int n, int kk, int f32) {
        dim3 grid((n + GBN - 1) / GBN, m / GBM);
        gemm_bias_kernel<<<grid, dim3(256), 0, stream>>>(A, Wt, bias, C, m, n, kk, f32);
    };

    // prep: bf16 activations + transposed bf16 weights
    cvt(seq, seqb, (long)M * Dm);
    trw(W_dq,  Wt_dq,  Dm,  Dc1);
    trw(W_dkv, Wt_dkv, Dm,  Dc);
    trw(W_rk,  Wt_rk,  Dm,  Drot);
    trw(W_uq,  Wt_uq,  Dc1, Dm);
    trw(W_rq,  Wt_rq,  Dc1, Hc * Drot);
    trw(W_uk,  Wt_uk,  Dc,  Dm);
    trw(W_uv,  Wt_uv,  Dc,  Dm);
    trw(W_o,   Wt_o,   Dm,  Dm);

    // projections (bf16 out)
    gemm(seqb, Wt_dq,  b_dq,  c_q,   M, Dc1,       Dm,  0);
    gemm(seqb, Wt_dkv, b_dkv, c_kv,  M, Dc,        Dm,  0);
    gemm(seqb, Wt_rk,  b_rk,  k_rot, M, Drot,      Dm,  0);
    gemm(c_q,  Wt_uq,  b_uq,  q,     M, Dm,        Dc1, 0);
    gemm(c_q,  Wt_rq,  b_rq,  q_rot, M, Hc * Drot, Dc1, 0);
    gemm(c_kv, Wt_uk,  b_uk,  k,     M, Dm,        Dc,  0);
    gemm(c_kv, Wt_uv,  b_uv,  vv,    M, Dm,        Dc,  0);

    // RoPE
    {
        long np = (long)M * Hc * 32;
        rope_kernel<<<(int)((np + 255) / 256), 256, 0, stream>>>(q_rot, np, Sc, Hc * 32);
    }
    {
        long np = (long)M * 32;
        rope_kernel<<<(int)((np + 255) / 256), 256, 0, stream>>>(k_rot, np, Sc, 32);
    }

    // pack q_full / k_full / v^T
    {
        long tot = (long)Bc * Hc * Sc * 192;
        int nb = (int)((tot + 255) / 256);
        pack_qfull_kernel<<<nb, 256, 0, stream>>>(q, q_rot, qfull, Sc, Hc, tot);
        pack_kfull_kernel<<<nb, 256, 0, stream>>>(k, k_rot, kfull, Sc, Hc, tot);
    }
    {
        long tot = (long)Bc * Hc * 128 * Sc;
        pack_vt_kernel<<<(int)((tot + 255) / 256), 256, 0, stream>>>(vv, vt, Sc, Hc, tot);
    }

    // attention
    {
        dim3 grid(Sc / 128, Hc, Bc);
        mla_flash_attn<<<grid, dim3(256), 0, stream>>>(qfull, kfull, vt, attn, Sc, Hc);
    }

    // output projection -> d_out (fp32)
    gemm(attn, Wt_o, b_o, d_out, M, Dm, Dm, 1);
}
